// CIE_89893665505337
// MI455X (gfx1250) — matrix operand order repaired and checked by reading, not compiled, not run on hardware
//
#include <hip/hip_runtime.h>

#define N_  4096
#define D_  128
#define H_  2
#define DH_ 64
#define FF_ 256
#define NE_ 2048
#define NNZ_ 65536

typedef __attribute__((ext_vector_type(16))) __bf16 v16bf;
typedef __attribute__((ext_vector_type(8)))  float  v8f;
typedef __attribute__((ext_vector_type(8)))  unsigned short u16x8;

union BFrag {
  v16bf v;
  unsigned short u[16];
  u16x8 h[2];          // two 16-byte halves -> 2x ds_load_b128
};

__device__ __forceinline__ unsigned short f2bf(float f) {
  unsigned int u = __float_as_uint(f);
  u += 0x7fffu + ((u >> 16) & 1u);   // round-to-nearest-even
  return (unsigned short)(u >> 16);
}

// ---------------------------------------------------------------------------
// Generic GEMM: C[M x Nc] = act(A[M x K] @ B[K x Nc] + bias)
// Block: 256 threads (8 waves). Tile: 128x64, BK=32. wave w -> rows 16w..16w+15.
// LDS: A row-major [row][k]; B transposed [col][k] so all fragment reads are
// contiguous 16B chunks (ds_load_b128).
// ACT: 0 = none, 1 = sigmoid
// ---------------------------------------------------------------------------
template <int ACT>
__global__ __launch_bounds__(256) void hgt_gemm_wmma(
    const float* __restrict__ A, const float* __restrict__ B,
    const float* __restrict__ bias, float* __restrict__ C,
    int M, int K, int Nc) {
  __shared__ unsigned short As[128][40];   // 80B row stride (16B aligned)
  __shared__ unsigned short Bt[64][40];    // [col][k]
  const int tid  = threadIdx.x;
  const int wave = tid >> 5;
  const int lane = tid & 31;
  const int lrow = lane & 15;
  const int g    = lane >> 4;
  const int row0 = blockIdx.y * 128;
  const int col0 = blockIdx.x * 64;

  v8f acc[4] = {};

  for (int k0 = 0; k0 < K; k0 += 32) {
    __syncthreads();
#pragma unroll
    for (int i = 0; i < 4; i++) {               // A tile 128x32 (1024 float4)
      int idx = tid + i * 256;
      int r = idx >> 3, c4 = (idx & 7) << 2;
      float4 f = *reinterpret_cast<const float4*>(&A[(size_t)(row0 + r) * K + k0 + c4]);
      As[r][c4 + 0] = f2bf(f.x); As[r][c4 + 1] = f2bf(f.y);
      As[r][c4 + 2] = f2bf(f.z); As[r][c4 + 3] = f2bf(f.w);
    }
#pragma unroll
    for (int i = 0; i < 2; i++) {               // B tile 32x64 -> transposed
      int idx = tid + i * 256;
      int r = idx >> 4, c4 = (idx & 15) << 2;
      float4 f = *reinterpret_cast<const float4*>(&B[(size_t)(k0 + r) * Nc + col0 + c4]);
      Bt[c4 + 0][r] = f2bf(f.x); Bt[c4 + 1][r] = f2bf(f.y);
      Bt[c4 + 2][r] = f2bf(f.z); Bt[c4 + 3][r] = f2bf(f.w);
    }
    __syncthreads();

    BFrag af;
    af.h[0] = *reinterpret_cast<const u16x8*>(&As[wave * 16 + lrow][g * 8]);
    af.h[1] = *reinterpret_cast<const u16x8*>(&As[wave * 16 + lrow][16 + g * 8]);
#pragma unroll
    for (int ct = 0; ct < 4; ct++) {
      BFrag bf;
      bf.h[0] = *reinterpret_cast<const u16x8*>(&Bt[ct * 16 + lrow][g * 8]);
      bf.h[1] = *reinterpret_cast<const u16x8*>(&Bt[ct * 16 + lrow][16 + g * 8]);
      acc[ct] = __builtin_amdgcn_wmma_f32_16x16x32_bf16(
          false, af.v, false, bf.v, (short)0, acc[ct], false, false);
    }
  }

#pragma unroll
  for (int ct = 0; ct < 4; ct++) {
#pragma unroll
    for (int r = 0; r < 8; r++) {
      int row = row0 + wave * 16 + r + g * 8;
      int col = col0 + ct * 16 + lrow;
      float v = acc[ct][r];
      if (bias) v += bias[col];
      if (ACT == 1) v = 1.0f / (1.0f + __expf(-v));
      C[(size_t)row * Nc + col] = v;
    }
  }
}

// ---------------------------------------------------------------------------
// Flash attention: one block = 128 query rows x one head. 8 waves, 16 rows each.
// K staged row-major [key][d] (B-frags of Q@K^T read contiguous d).
// V staged transposed [d][key] (B-frags of P@V read contiguous key).
// P round-trips through wave-private LDS row-major (A-frags read contiguous k).
// ---------------------------------------------------------------------------
__global__ __launch_bounds__(256) void hgt_attn_wmma(
    const float* __restrict__ Q, const float* __restrict__ Km,
    const float* __restrict__ Vm, float* __restrict__ O) {
  __shared__ unsigned short KsS[64][72];      // [key][d], 144B stride
  __shared__ unsigned short VtS[64][72];      // [d][key]
  __shared__ unsigned short PsS[8][16][72];   // wave-private P

  const int tid  = threadIdx.x;
  const int wave = tid >> 5;
  const int lane = tid & 31;
  const int lrow = lane & 15;
  const int g    = lane >> 4;
  const int q0   = blockIdx.x * 128;
  const int dc0  = blockIdx.y * DH_;          // head column offset

  // Q fragments (pre-scaled by 1/sqrt(DH)=0.125), wide global loads
  BFrag qf[2];
  const int qrow = q0 + wave * 16 + lrow;
#pragma unroll
  for (int ks = 0; ks < 2; ks++) {
#pragma unroll
    for (int c = 0; c < 2; c++) {             // chunk c: K base ks*32 + c*16 + g*8
      const float* p = &Q[(size_t)qrow * D_ + dc0 + ks * 32 + c * 16 + g * 8];
      float4 f0 = *reinterpret_cast<const float4*>(p);
      float4 f1 = *reinterpret_cast<const float4*>(p + 4);
      int b = c * 8;
      qf[ks].u[b + 0] = f2bf(f0.x * 0.125f); qf[ks].u[b + 1] = f2bf(f0.y * 0.125f);
      qf[ks].u[b + 2] = f2bf(f0.z * 0.125f); qf[ks].u[b + 3] = f2bf(f0.w * 0.125f);
      qf[ks].u[b + 4] = f2bf(f1.x * 0.125f); qf[ks].u[b + 5] = f2bf(f1.y * 0.125f);
      qf[ks].u[b + 6] = f2bf(f1.z * 0.125f); qf[ks].u[b + 7] = f2bf(f1.w * 0.125f);
    }
  }

  float mi[8], li[8];
#pragma unroll
  for (int r = 0; r < 8; r++) { mi[r] = -1e30f; li[r] = 0.0f; }
  v8f of4[4] = {};

  for (int j0 = 0; j0 < N_; j0 += 64) {
    __syncthreads();
#pragma unroll
    for (int i = 0; i < 4; i++) {             // stage 64x64 K (as-is) and V (transposed)
      int idx = tid + i * 256;
      int r = idx >> 4, c4 = (idx & 15) << 2;
      float4 fk = *reinterpret_cast<const float4*>(&Km[(size_t)(j0 + r) * D_ + dc0 + c4]);
      KsS[r][c4 + 0] = f2bf(fk.x); KsS[r][c4 + 1] = f2bf(fk.y);
      KsS[r][c4 + 2] = f2bf(fk.z); KsS[r][c4 + 3] = f2bf(fk.w);
      float4 fv = *reinterpret_cast<const float4*>(&Vm[(size_t)(j0 + r) * D_ + dc0 + c4]);
      VtS[c4 + 0][r] = f2bf(fv.x); VtS[c4 + 1][r] = f2bf(fv.y);
      VtS[c4 + 2][r] = f2bf(fv.z); VtS[c4 + 3][r] = f2bf(fv.w);
    }
    __syncthreads();

    // S = Q @ K^T : 16 x 64 per wave
    v8f sa[4] = {};
#pragma unroll
    for (int ct = 0; ct < 4; ct++) {
#pragma unroll
      for (int ks = 0; ks < 2; ks++) {
        BFrag bf;
        bf.h[0] = *reinterpret_cast<const u16x8*>(&KsS[ct * 16 + lrow][ks * 32 + g * 8]);
        bf.h[1] = *reinterpret_cast<const u16x8*>(&KsS[ct * 16 + lrow][ks * 32 + 16 + g * 8]);
        sa[ct] = __builtin_amdgcn_wmma_f32_16x16x32_bf16(
            false, qf[ks].v, false, bf.v, (short)0, sa[ct], false, false);
      }
    }

    // Online softmax (row = r + 8g lives across the 16 lanes of one half-wave)
    float alpha[8], rsum[8];
#pragma unroll
    for (int r = 0; r < 8; r++) {
      float m = -1e30f;
#pragma unroll
      for (int ct = 0; ct < 4; ct++) m = fmaxf(m, sa[ct][r]);
      m = fmaxf(m, __shfl_xor(m, 8, 32));
      m = fmaxf(m, __shfl_xor(m, 4, 32));
      m = fmaxf(m, __shfl_xor(m, 2, 32));
      m = fmaxf(m, __shfl_xor(m, 1, 32));
      float mn = fmaxf(mi[r], m);
      alpha[r] = __expf(mi[r] - mn);
      mi[r] = mn;
      rsum[r] = 0.0f;
    }
#pragma unroll
    for (int ct = 0; ct < 4; ct++) {
#pragma unroll
      for (int r = 0; r < 8; r++) {
        float p = __expf(sa[ct][r] - mi[r]);
        rsum[r] += p;
        PsS[wave][r + 8 * g][ct * 16 + lrow] = f2bf(p);
      }
    }
#pragma unroll
    for (int r = 0; r < 8; r++) {
      float s = rsum[r];
      s += __shfl_xor(s, 8, 32);
      s += __shfl_xor(s, 4, 32);
      s += __shfl_xor(s, 2, 32);
      s += __shfl_xor(s, 1, 32);
      li[r] = li[r] * alpha[r] + s;
    }
#pragma unroll
    for (int ct = 0; ct < 4; ct++)
#pragma unroll
      for (int r = 0; r < 8; r++) of4[ct][r] *= alpha[r];

    // Reload P as A-fragments (contiguous 16B chunks; DS in-order per wave)
    BFrag pf[2];
#pragma unroll
    for (int ks = 0; ks < 2; ks++) {
      pf[ks].h[0] = *reinterpret_cast<const u16x8*>(&PsS[wave][lrow][ks * 32 + g * 8]);
      pf[ks].h[1] = *reinterpret_cast<const u16x8*>(&PsS[wave][lrow][ks * 32 + 16 + g * 8]);
    }

    // O += P @ V
#pragma unroll
    for (int ct = 0; ct < 4; ct++) {
#pragma unroll
      for (int ks = 0; ks < 2; ks++) {
        BFrag bf;
        bf.h[0] = *reinterpret_cast<const u16x8*>(&VtS[ct * 16 + lrow][ks * 32 + g * 8]);
        bf.h[1] = *reinterpret_cast<const u16x8*>(&VtS[ct * 16 + lrow][ks * 32 + 16 + g * 8]);
        of4[ct] = __builtin_amdgcn_wmma_f32_16x16x32_bf16(
            false, pf[ks].v, false, bf.v, (short)0, of4[ct], false, false);
      }
    }
  }

#pragma unroll
  for (int ct = 0; ct < 4; ct++) {
#pragma unroll
    for (int r = 0; r < 8; r++) {
      int row = q0 + wave * 16 + r + 8 * g;
      int col = dc0 + ct * 16 + lrow;
      O[(size_t)row * D_ + col] = of4[ct][r] / li[r];
    }
  }
}

// ---------------------------------------------------------------------------
// LayerNorm(h + y) in place; one wave per 128-wide row, 8 rows per block.
// ---------------------------------------------------------------------------
__global__ __launch_bounds__(256) void hgt_ln_res(
    float* __restrict__ h, const float* __restrict__ y,
    const float* __restrict__ gam, const float* __restrict__ bet) {
  int wave = threadIdx.x >> 5, lane = threadIdx.x & 31;
  size_t row = (size_t)blockIdx.x * 8 + wave;
  float v[4], s = 0.0f, s2 = 0.0f;
#pragma unroll
  for (int i = 0; i < 4; i++) {
    int c = lane + i * 32;
    float t = h[row * D_ + c] + y[row * D_ + c];
    v[i] = t; s += t; s2 += t * t;
  }
#pragma unroll
  for (int off = 16; off >= 1; off >>= 1) {
    s  += __shfl_xor(s,  off, 32);
    s2 += __shfl_xor(s2, off, 32);
  }
  float mean = s * (1.0f / D_);
  float var  = s2 * (1.0f / D_) - mean * mean;
  float inv  = rsqrtf(var + 1e-5f);
#pragma unroll
  for (int i = 0; i < 4; i++) {
    int c = lane + i * 32;
    h[row * D_ + c] = (v[i] - mean) * inv * gam[c] + bet[c];
  }
}

__global__ void hgt_copy4(float* __restrict__ dst, const float* __restrict__ src) {
  size_t i = ((size_t)blockIdx.x * 256 + threadIdx.x) * 4;
  *reinterpret_cast<float4*>(dst + i) = *reinterpret_cast<const float4*>(src + i);
}

__global__ void hgt_zero(float* __restrict__ p, int n) {
  int i = blockIdx.x * 256 + threadIdx.x;
  if (i < n) p[i] = 0.0f;
}

__global__ void hgt_degrees(const int* __restrict__ node_idx, const int* __restrict__ he_idx,
                            float* __restrict__ ddeg, float* __restrict__ bdeg) {
  int i = blockIdx.x * 256 + threadIdx.x;
  atomicAdd(&bdeg[he_idx[i]], 1.0f);
  atomicAdd(&ddeg[node_idx[i]], 1.0f);
}

// dst[di[i]*D + d] += src[si[i]*D + d]   over all (i, d)
__global__ void hgt_scatter_add(float* __restrict__ dst, const float* __restrict__ src,
                                const int* __restrict__ di, const int* __restrict__ si) {
  size_t idx = (size_t)blockIdx.x * 256 + threadIdx.x;
  int i = (int)(idx >> 7);          // / D_
  int d = (int)(idx & (D_ - 1));
  atomicAdd(&dst[(size_t)di[i] * D_ + d], src[(size_t)si[i] * D_ + d]);
}

__global__ void hgt_scale_seg(float* __restrict__ e, const float* __restrict__ deg) {
  size_t idx = (size_t)blockIdx.x * 256 + threadIdx.x;
  float dg = deg[idx >> 7];
  e[idx] *= (dg > 0.0f) ? (1.0f / dg) : 0.0f;
}

__global__ void hgt_final(float* __restrict__ out, const float* __restrict__ acc,
                          const float* __restrict__ deg, const float* __restrict__ bh) {
  size_t idx = (size_t)blockIdx.x * 256 + threadIdx.x;
  float dg = deg[idx >> 7];
  float v = acc[idx] * ((dg > 0.0f) ? (1.0f / dg) : 0.0f) + bh[idx & (D_ - 1)];
  out[idx] = fmaxf(v, 0.0f);
}

// ---------------------------------------------------------------------------
extern "C" void kernel_launch(void* const* d_in, const int* in_sizes, int n_in,
                              void* d_out, int out_size, void* d_ws, size_t ws_size,
                              hipStream_t stream) {
  (void)in_sizes; (void)n_in; (void)out_size; (void)ws_size;
  const float* x    = (const float*)d_in[0];
  const int*   edge = (const int*)d_in[1];     // [node_idx(NNZ), he_idx(NNZ)]
  const float* Wq = (const float*)d_in[2];  const float* bq = (const float*)d_in[3];
  const float* Wk = (const float*)d_in[4];  const float* bk = (const float*)d_in[5];
  const float* Wv = (const float*)d_in[6];  const float* bv = (const float*)d_in[7];
  const float* Wo = (const float*)d_in[8];  const float* bo = (const float*)d_in[9];
  const float* g1 = (const float*)d_in[10]; const float* b1 = (const float*)d_in[11];
  const float* W1 = (const float*)d_in[12]; const float* bf1 = (const float*)d_in[13];
  const float* W2 = (const float*)d_in[14]; const float* bf2 = (const float*)d_in[15];
  const float* g2 = (const float*)d_in[16]; const float* b2 = (const float*)d_in[17];
  const float* Wh = (const float*)d_in[18]; const float* bh = (const float*)d_in[19];

  const size_t ND = (size_t)N_ * D_;
  float* ws = (float*)d_ws;
  float* Hb = ws;             // N*D  current activations
  float* Qb = ws + 1 * ND;    // N*D
  float* Kb = ws + 2 * ND;    // N*D  (later: hyperedge accum E, NE*D <= N*D)
  float* Vb = ws + 3 * ND;    // N*D  (later: Bdeg[NE] ++ Ddeg[N])
  float* Ob = ws + 4 * ND;    // N*D  (later: node accum)
  float* Zb = ws + 5 * ND;    // N*FF

  const int* node_idx = edge;
  const int* he_idx   = edge + NNZ_;

  dim3 blk(256);
  dim3 gDD(D_ / 64, N_ / 128);    // GEMM K->D
  dim3 gDF(FF_ / 64, N_ / 128);   // GEMM D->FF
  dim3 gAtt(N_ / 128, H_);

  hgt_copy4<<<ND / 1024, blk, 0, stream>>>(Hb, x);

  for (int layer = 0; layer < 2; layer++) {
    // MHA
    hgt_gemm_wmma<0><<<gDD, blk, 0, stream>>>(Hb, Wq, bq, Qb, N_, D_, D_);
    hgt_gemm_wmma<0><<<gDD, blk, 0, stream>>>(Hb, Wk, bk, Kb, N_, D_, D_);
    hgt_gemm_wmma<0><<<gDD, blk, 0, stream>>>(Hb, Wv, bv, Vb, N_, D_, D_);
    hgt_attn_wmma<<<gAtt, blk, 0, stream>>>(Qb, Kb, Vb, Ob);
    hgt_gemm_wmma<0><<<gDD, blk, 0, stream>>>(Ob, Wo, bo, Qb, N_, D_, D_);
    hgt_ln_res<<<N_ / 8, blk, 0, stream>>>(Hb, Qb, g1, b1);
    // FFN (sigmoid gate)
    hgt_gemm_wmma<1><<<gDF, blk, 0, stream>>>(Hb, W1, bf1, Zb, N_, D_, FF_);
    hgt_gemm_wmma<0><<<gDD, blk, 0, stream>>>(Zb, W2, bf2, Qb, N_, FF_, D_);
    hgt_ln_res<<<N_ / 8, blk, 0, stream>>>(Hb, Qb, g2, b2);
  }

  // HypergraphConv: xt = h @ Wh (no bias yet)
  hgt_gemm_wmma<0><<<gDD, blk, 0, stream>>>(Hb, Wh, nullptr, Qb, N_, D_, D_);

  float* Eb   = Kb;            // NE*D
  float* Bdeg = Vb;            // NE
  float* Ddeg = Vb + NE_;      // N
  float* Acc  = Ob;            // N*D

  hgt_zero<<<(NE_ * D_ + 255) / 256, blk, 0, stream>>>(Eb, NE_ * D_);
  hgt_zero<<<((int)ND + 255) / 256, blk, 0, stream>>>(Acc, (int)ND);
  hgt_zero<<<(NE_ + N_ + 255) / 256, blk, 0, stream>>>(Bdeg, NE_ + N_);

  hgt_degrees<<<NNZ_ / 256, blk, 0, stream>>>(node_idx, he_idx, Ddeg, Bdeg);
  // e = segsum(xt[node]) by hyperedge
  hgt_scatter_add<<<(unsigned)((size_t)NNZ_ * D_ / 256), blk, 0, stream>>>(Eb, Qb, he_idx, node_idx);
  hgt_scale_seg<<<NE_ * D_ / 256, blk, 0, stream>>>(Eb, Bdeg);
  // acc = segsum(e[he]) by node
  hgt_scatter_add<<<(unsigned)((size_t)NNZ_ * D_ / 256), blk, 0, stream>>>(Acc, Eb, node_idx, he_idx);
  hgt_final<<<(unsigned)(ND / 256), blk, 0, stream>>>((float*)d_out, Acc, Ddeg, bh);
}
